// PairAwareBlock_82308753261000
// MI455X (gfx1250) — matrix operand order repaired and checked by reading, not compiled, not run on hardware
//
#include <hip/hip_runtime.h>
#include <math.h>

typedef __bf16 bf16;
typedef __attribute__((ext_vector_type(16))) __bf16 v16bf;
typedef __attribute__((ext_vector_type(8)))  __bf16 v8bf;
typedef __attribute__((ext_vector_type(8)))  float  v8f;

#define LSEQ 256
#define LL   65536   // 256*256

enum {
  F_BIAS = 1, F_RELU = 2, F_SIG = 4, F_GATE = 8,
  F_ROWSCALE = 16, F_RES = 32, F_OUTF = 64, F_OUTB = 128
};

// ---------------------------------------------------------------------------
// WMMA helper: D = A(16x32 bf16) x B(32x16 bf16)^T-form + C (fp32)
// ---------------------------------------------------------------------------
__device__ __forceinline__ v8f wmma_bf16(v16bf a, v16bf b, v8f c) {
#if defined(__HIP_DEVICE_COMPILE__)
  return __builtin_amdgcn_wmma_f32_16x16x32_bf16(false, a, false, b, (short)0, c,
                                                 false, false);
#else
  return c;
#endif
}

__device__ __forceinline__ v16bf make_a_frag(const bf16* __restrict__ rowp, int lane) {
  int h16 = (lane >> 4) * 16;
  v8bf lo = *(const v8bf*)(rowp + h16);
  v8bf hi = *(const v8bf*)(rowp + h16 + 8);
  v16bf r;
#pragma unroll
  for (int i = 0; i < 8; ++i) { r[i] = lo[i]; r[8 + i] = hi[i]; }
  return r;
}

__device__ __forceinline__ void wait_asynccnt0() {
#if defined(__HIP_DEVICE_COMPILE__)
#if __has_builtin(__builtin_amdgcn_s_wait_asynccnt)
  __builtin_amdgcn_s_wait_asynccnt(0);
#else
  asm volatile("s_wait_asynccnt 0x0" ::: "memory");
#endif
#endif
}

// Issue one 16B async global->LDS copy (ASYNCcnt tracked, no VGPR data).
__device__ __forceinline__ void async_copy_b128(const bf16* g, bf16* lds_ptr) {
#if defined(__HIP_DEVICE_COMPILE__)
  unsigned lo = (unsigned)(uintptr_t)lds_ptr;          // workgroup LDS offset
  unsigned long long ga = (unsigned long long)(uintptr_t)g;
  asm volatile("global_load_async_to_lds_b128 %0, %1, off"
               :: "v"(lo), "v"(ga) : "memory");
#else
  (void)g; (void)lds_ptr;
#endif
}

// Stage a 64(cols) x 32(k) bf16 W tile into LDS: 256 chunks of 16B, 2/thread.
__device__ __forceinline__ void stage_w_async(const bf16* __restrict__ W, int n0,
                                              int K, int k0, bf16* ws, int tid) {
#pragma unroll
  for (int q = 0; q < 2; ++q) {
    int c   = tid * 2 + q;          // chunk 0..255
    int col = c >> 2;               // local col 0..63
    int off = (c & 3) * 8;          // k offset within 32
    async_copy_b128(W + (size_t)(n0 + col) * K + k0 + off, ws + c * 8);
  }
}

// ---------------------------------------------------------------------------
// Full-tile NT GEMM (N % 64 == 0, M % 64 == 0, K % 32 == 0):
//   C[M,N] = epilogue( A[M,K](bf16) * W[N,K](bf16)^T )
// Block = 128 threads (4 waves), tile 64x64. W tile double-buffered in LDS via
// GLOBAL_LOAD_ASYNC_TO_LDS_B128 (4x cross-wave reuse), overlapped with WMMA.
// ---------------------------------------------------------------------------
__global__ __launch_bounds__(128) void k_gemm_nt_full(
    const bf16* __restrict__ A, const bf16* __restrict__ W,
    int M, int N, int K,
    const float* __restrict__ bias, const float* __restrict__ gate,
    const float* __restrict__ rowscale, const float* __restrict__ res,
    float* __restrict__ outf, bf16* __restrict__ outb, int flags) {
  __shared__ __attribute__((aligned(16))) bf16 WS[2][2048];  // 2 x 64x32

  int tid  = threadIdx.x;
  int lane = tid & 31;
  int w    = tid >> 5;
  int m0   = blockIdx.y * 64 + w * 16;
  int n0   = blockIdx.x * 64;
  int h16  = (lane >> 4) * 16;
  int h8   = (lane >> 4) * 8;
  const bf16* arow = A + (size_t)(m0 + (lane & 15)) * K;

  v8f acc[4] = {};
  int nk  = K >> 5;
  int buf = 0;
  stage_w_async(W, n0, K, 0, WS[0], tid);

  for (int kk = 0; kk < nk; ++kk) {
    int k0 = kk * 32;
    wait_asynccnt0();        // own async copies complete
    __syncthreads();         // staged tile visible to all waves
    if (kk + 1 < nk) stage_w_async(W, n0, K, k0 + 32, WS[buf ^ 1], tid);
    __builtin_prefetch(arow + k0 + 256, 0, 1);  // global_prefetch_b8
    v16bf af = make_a_frag(arow + k0, lane);
    const bf16* wsb = WS[buf];
#pragma unroll
    for (int t = 0; t < 4; ++t) {
      int coll = t * 16 + (lane & 15);
      v16bf bfr = *(const v16bf*)(wsb + coll * 32 + h16);  // ds_load from LDS
      acc[t] = wmma_bf16(af, bfr, acc[t]);
    }
    __syncthreads();         // all reads of buf done before it is re-staged
    buf ^= 1;
  }

#pragma unroll
  for (int t = 0; t < 4; ++t) {
    int col = n0 + t * 16 + (lane & 15);
    float bval = (flags & F_BIAS) ? bias[col] : 0.0f;
#pragma unroll
    for (int r = 0; r < 8; ++r) {
      int m = m0 + h8 + r;
      size_t idx = (size_t)m * N + col;
      float v = acc[t][r] + bval;
      if (flags & F_RELU) v = fmaxf(v, 0.0f);
      if (flags & F_SIG)  v = 1.0f / (1.0f + __expf(-v));
      if (flags & F_GATE) v *= gate[idx];
      if (flags & F_ROWSCALE) v *= rowscale[m];
      if (flags & F_RES)  v += res[idx];
      if (flags & F_OUTF) outf[idx] = v;
      if (flags & F_OUTB) outb[idx] = (bf16)v;
    }
  }
}

// ---------------------------------------------------------------------------
// Guarded NT GEMM for ragged N (N=12 pair->seq bias, N=4 triangle bias).
// ---------------------------------------------------------------------------
__global__ __launch_bounds__(128) void k_gemm_nt(
    const bf16* __restrict__ A, const bf16* __restrict__ W,
    int M, int N, int K,
    const float* __restrict__ bias, const float* __restrict__ gate,
    const float* __restrict__ rowscale, const float* __restrict__ res,
    float* __restrict__ outf, bf16* __restrict__ outb, int flags) {
  int lane = threadIdx.x & 31;
  int w    = threadIdx.x >> 5;
  int m0   = blockIdx.y * 64 + w * 16;
  int n0   = blockIdx.x * 64;
  if (m0 >= M) return;

  int h8  = (lane >> 4) * 8;
  int h16 = (lane >> 4) * 16;
  const bf16* arow = A + (size_t)(m0 + (lane & 15)) * K;

  v8f acc[4] = {};
  for (int k0 = 0; k0 < K; k0 += 32) {
    v16bf af = make_a_frag(arow + k0, lane);
#pragma unroll
    for (int t = 0; t < 4; ++t) {
      int col = n0 + t * 16 + (lane & 15);
      v16bf bfr;
      if (col < N) {
        bfr = *(const v16bf*)(W + (size_t)col * K + k0 + h16);
      } else {
#pragma unroll
        for (int q = 0; q < 16; ++q) bfr[q] = (bf16)0.0f;
      }
      acc[t] = wmma_bf16(af, bfr, acc[t]);
    }
  }

#pragma unroll
  for (int t = 0; t < 4; ++t) {
    int col = n0 + t * 16 + (lane & 15);
    if (col >= N) continue;
    float bval = (flags & F_BIAS) ? bias[col] : 0.0f;
#pragma unroll
    for (int r = 0; r < 8; ++r) {
      int m = m0 + h8 + r;
      size_t idx = (size_t)m * N + col;
      float v = acc[t][r] + bval;
      if (flags & F_RELU) v = fmaxf(v, 0.0f);
      if (flags & F_SIG)  v = 1.0f / (1.0f + __expf(-v));
      if (flags & F_GATE) v *= gate[idx];
      if (flags & F_ROWSCALE) v *= rowscale[m];
      if (flags & F_RES)  v += res[idx];
      if (flags & F_OUTF) outf[idx] = v;
      if (flags & F_OUTB) outb[idx] = (bf16)v;
    }
  }
}

// ---------------------------------------------------------------------------
// Batched triangle einsum: X[m,col,d] = sum_k Apk[d][m][k] * Bpk[d][col][k]
// (per-d 256x256x256 NT GEMM; 128 d-channels on blockIdx.z)
// ---------------------------------------------------------------------------
__global__ __launch_bounds__(128) void k_einsum_nt(
    const bf16* __restrict__ Apk, const bf16* __restrict__ Bpk,
    float* __restrict__ X) {
  int d = blockIdx.z;
  const bf16* A = Apk + (size_t)d * 65536;
  const bf16* B = Bpk + (size_t)d * 65536;
  int lane = threadIdx.x & 31;
  int w    = threadIdx.x >> 5;
  int m0   = blockIdx.y * 64 + w * 16;
  int n0   = blockIdx.x * 64;
  int h8   = (lane >> 4) * 8;
  int h16  = (lane >> 4) * 16;
  const bf16* arow = A + (size_t)(m0 + (lane & 15)) * 256;

  v8f acc[4] = {};
  for (int k0 = 0; k0 < 256; k0 += 32) {
    __builtin_prefetch(arow + k0 + 256, 0, 1);
    v16bf af = make_a_frag(arow + k0, lane);
#pragma unroll
    for (int t = 0; t < 4; ++t) {
      int col = n0 + t * 16 + (lane & 15);
      v16bf bfr = *(const v16bf*)(B + (size_t)col * 256 + k0 + h16);
      acc[t] = wmma_bf16(af, bfr, acc[t]);
    }
  }
#pragma unroll
  for (int t = 0; t < 4; ++t) {
    int col = n0 + t * 16 + (lane & 15);
#pragma unroll
    for (int r = 0; r < 8; ++r) {
      int m = m0 + h8 + r;
      X[((size_t)m * 256 + col) * 128 + d] = acc[t][r];
    }
  }
}

// ---------------------------------------------------------------------------
// Fused attention over one (batch, j-tile of 32 rows):
//   S = scale*Q K^T + bias + addk ; P = softmax_k(S) ; Y = sigmoid(gate)*(P V)
// Block = 64 threads (2 waves). grid = (Lq/32, nbatch), batch = i*nheads + h.
// ---------------------------------------------------------------------------
__global__ __launch_bounds__(64) void k_attn(
    const bf16* __restrict__ Q, const bf16* __restrict__ Kv,
    const bf16* __restrict__ Vt,
    const float* __restrict__ bias, int biasH,
    const float* __restrict__ gate,
    const float* __restrict__ addk, int addkStride,
    bf16* __restrict__ Yout, int nheads, int rowStride, float scale) {
  __shared__ float S[32][256];
  __shared__ __attribute__((aligned(16))) bf16 Pm[32][256];

  int b    = blockIdx.y;
  int h    = b % nheads;
  int ib   = b / nheads;
  int j0   = blockIdx.x * 32;
  int lane = threadIdx.x & 31;
  int w    = threadIdx.x >> 5;
  int h8   = (lane >> 4) * 8;
  int h16  = (lane >> 4) * 16;

  size_t base = (size_t)ib * 256 * rowStride + (size_t)h * 32;

  // --- Phase 1: scores (16 column tiles; K(contraction)=32 -> 1 wmma each)
  const bf16* qrow = Q + base + (size_t)(j0 + w * 16 + (lane & 15)) * rowStride;
  v16bf qf = make_a_frag(qrow, lane);
  const float* addkp = addk + (size_t)ib * addkStride;

  for (int kt = 0; kt < 16; ++kt) {
    int k0   = kt * 16;
    int kcol = k0 + (lane & 15);
    v16bf kf = *(const v16bf*)(Kv + base + (size_t)kcol * rowStride + h16);
    v8f acc = {};
    acc = wmma_bf16(qf, kf, acc);
#pragma unroll
    for (int r = 0; r < 8; ++r) {
      int jl = w * 16 + h8 + r;
      int jg = j0 + jl;
      float v = acc[r] * scale + bias[((size_t)jg * 256 + kcol) * biasH + h] +
                addkp[kcol];
      S[jl][kcol] = v;
    }
  }
  __syncthreads();

  // --- Phase 2: row softmax (one row per thread; 32 rows)
  if (threadIdx.x < 32) {
    int r = threadIdx.x;
    float mx = -3.4e38f;
    for (int k = 0; k < 256; ++k) mx = fmaxf(mx, S[r][k]);
    float sum = 0.0f;
    for (int k = 0; k < 256; ++k) { float e = __expf(S[r][k] - mx); S[r][k] = e; sum += e; }
    float inv = 1.0f / sum;
    for (int k = 0; k < 256; ++k) Pm[r][k] = (bf16)(S[r][k] * inv);
  }
  __syncthreads();

  // --- Phase 3: Y = P * V  (P from LDS via ds loads, Vt[c][k] from global)
  const bf16* vtb = Vt + ((size_t)(ib * nheads + h) * 32) * 256;
  v8f acc0 = {}, acc1 = {};
  int jl = w * 16 + (lane & 15);
  for (int kt = 0; kt < 8; ++kt) {
    int k0 = kt * 32;
    v8bf lo = *(const v8bf*)&Pm[jl][k0 + h16];
    v8bf hi = *(const v8bf*)&Pm[jl][k0 + h16 + 8];
    v16bf pf;
#pragma unroll
    for (int t = 0; t < 8; ++t) { pf[t] = lo[t]; pf[8 + t] = hi[t]; }
    v16bf v0 = *(const v16bf*)(vtb + (size_t)(0 + (lane & 15)) * 256 + k0 + h16);
    v16bf v1 = *(const v16bf*)(vtb + (size_t)(16 + (lane & 15)) * 256 + k0 + h16);
    acc0 = wmma_bf16(pf, v0, acc0);
    acc1 = wmma_bf16(pf, v1, acc1);
  }
#pragma unroll
  for (int r = 0; r < 8; ++r) {
    int jg = j0 + w * 16 + h8 + r;
    size_t rowoff = (size_t)ib * 256 * rowStride + (size_t)jg * rowStride + h * 32;
    int c0 = (lane & 15);
    float g0 = gate[rowoff + c0];
    Yout[rowoff + c0] = (bf16)(acc0[r] * g0);
    int c1 = 16 + (lane & 15);
    float g1 = gate[rowoff + c1];
    Yout[rowoff + c1] = (bf16)(acc1[r] * g1);
  }
}

// ---------------------------------------------------------------------------
// LayerNorm over last dim D; writes bf16. transposeL: write row (j*256+i).
// ---------------------------------------------------------------------------
__global__ __launch_bounds__(128) void k_layernorm(
    const float* __restrict__ x, const float* __restrict__ g,
    const float* __restrict__ b, bf16* __restrict__ out, int D, int transposeL) {
  __shared__ float s1[128], s2[128];
  int rowi = blockIdx.x;
  const float* xr = x + (size_t)rowi * D;
  float a = 0.0f, a2 = 0.0f;
  for (int i = threadIdx.x; i < D; i += 128) { float v = xr[i]; a += v; a2 += v * v; }
  s1[threadIdx.x] = a; s2[threadIdx.x] = a2;
  __syncthreads();
  for (int s = 64; s > 0; s >>= 1) {
    if (threadIdx.x < (unsigned)s) {
      s1[threadIdx.x] += s1[threadIdx.x + s];
      s2[threadIdx.x] += s2[threadIdx.x + s];
    }
    __syncthreads();
  }
  float mean = s1[0] / D;
  float var  = s2[0] / D - mean * mean;
  float inv  = rsqrtf(var + 1e-5f);
  size_t orow = rowi;
  if (transposeL) orow = (size_t)(rowi & 255) * 256 + (rowi >> 8);
  bf16* op = out + orow * D;
  for (int i = threadIdx.x; i < D; i += 128)
    op[i] = (bf16)((xr[i] - mean) * inv * g[i] + b[i]);
}

// ---------------------------------------------------------------------------
// Small utility kernels
// ---------------------------------------------------------------------------
__global__ void k_cast(const float* __restrict__ s, bf16* __restrict__ d, int n) {
  int t = blockIdx.x * 256 + threadIdx.x;
  if (t < n) d[t] = (bf16)s[t];
}

__global__ void k_prep_mask(const unsigned char* __restrict__ mask,
                            float* __restrict__ madd_seq, float* __restrict__ pm,
                            float* __restrict__ madd_tri) {
  int t = blockIdx.x * 256 + threadIdx.x;  // 65536 threads
  int i = t >> 8, k = t & 255;
  float mi = mask[i] ? 1.0f : 0.0f;
  float mk = mask[k] ? 1.0f : 0.0f;
  if (i == 0) madd_seq[k] = (mk != 0.0f) ? 0.0f : -1e9f;
  float p = mi * mk;
  pm[t] = p;
  madd_tri[t] = 1e9f * (p - 1.0f);
}

// dst[d][i][k] (bf16) from src[(i*256+k),d] (outgoing) or src[(k*256+i),d] (incoming)
__global__ void k_pack_dlk(const float* __restrict__ src, bf16* __restrict__ dst,
                           int transpose) {
  size_t t = (size_t)blockIdx.x * 256 + threadIdx.x;  // 128*256*256
  int k = (int)(t & 255), i = (int)((t >> 8) & 255), d = (int)(t >> 16);
  size_t si = transpose ? ((size_t)(k * 256 + i) * 128 + d)
                        : ((size_t)(i * 256 + k) * 128 + d);
  dst[t] = (bf16)src[si];
}

// vt[(i*nheads+h)*32 + c][k] = v[(i*256+k)*ld + h*32 + c]
__global__ void k_pack_vt(const float* __restrict__ v, bf16* __restrict__ vt,
                          int nheads, int ld) {
  size_t t = (size_t)blockIdx.x * 256 + threadIdx.x;
  int k = (int)(t & 255);
  size_t row = t >> 8;
  int c  = (int)(row & 31);
  int hh = (int)((row >> 5) % nheads);
  int ib = (int)(row / (32 * (size_t)nheads));
  vt[t] = (bf16)v[((size_t)ib * 256 + k) * ld + hh * 32 + c];
}

// seq->pair outer: dst[(i*256+j), c] : c<64 -> q[j,c]*k[i,c] ; else q[j]-k[i]
__global__ void k_s2p_outer(const float* __restrict__ s, bf16* __restrict__ dst) {
  size_t t = (size_t)blockIdx.x * 256 + threadIdx.x;  // LL*128
  int c = (int)(t & 127);
  size_t m = t >> 7;
  int i = (int)(m >> 8), j = (int)(m & 255);
  int cc = (c < 64) ? c : (c - 64);
  float q = s[(size_t)j * 128 + cc];
  float kk = s[(size_t)i * 128 + 64 + cc];
  dst[t] = (bf16)((c < 64) ? (q * kk) : (q - kk));
}

// pair[(i*256+j),d] += y[(j*256+i),d]
__global__ void k_add_transposed(float* __restrict__ pair,
                                 const float* __restrict__ y) {
  size_t t = (size_t)blockIdx.x * 256 + threadIdx.x;  // LL*128
  int d = (int)(t & 127);
  size_t m = t >> 7;
  int i = (int)(m >> 8), j = (int)(m & 255);
  pair[t] += y[((size_t)j * 256 + i) * 128 + d];
}

// ---------------------------------------------------------------------------
// Host orchestration
// ---------------------------------------------------------------------------
extern "C" void kernel_launch(void* const* d_in, const int* in_sizes, int n_in,
                              void* d_out, int out_size, void* d_ws, size_t ws_size,
                              hipStream_t stream) {
  (void)in_sizes; (void)n_in; (void)out_size; (void)ws_size;

  const float* in_seq  = (const float*)d_in[0];
  const float* in_pair = (const float*)d_in[1];
  const unsigned char* in_mask = (const unsigned char*)d_in[2];
  int pi = 3;
  auto P = [&]() { return (const float*)d_in[pi++]; };

  const float* ln_seq_g = P(); const float* ln_seq_b = P();
  const float* p2s_ln_g = P(); const float* p2s_ln_b = P(); const float* p2s_w = P();
  const float* a_wq = P(); const float* a_wk = P(); const float* a_wv = P();
  const float* a_wo = P(); const float* a_wg = P(); const float* a_bg = P();
  const float* ts_ln_g = P(); const float* ts_ln_b = P();
  const float* ts_w1 = P(); const float* ts_b1 = P();
  const float* ts_w2 = P(); const float* ts_b2 = P();
  const float* sp_ln_g = P(); const float* sp_ln_b = P();
  const float* sp_wp = P(); const float* sp_bp = P();
  const float* sp_wo = P(); const float* sp_bo = P();

  struct TM {
    const float *ln_in_g, *ln_in_b, *ag_w, *ag_b, *ap_w, *ap_b;
    const float *bg_w, *bg_b, *bp_w, *bp_b, *ln_out_g, *ln_out_b;
    const float *wz, *bz, *g_w, *g_b;
  } tmo, tmi;
  auto loadTM = [&](TM& t) {
    t.ln_in_g = P(); t.ln_in_b = P(); t.ag_w = P(); t.ag_b = P();
    t.ap_w = P(); t.ap_b = P(); t.bg_w = P(); t.bg_b = P();
    t.bp_w = P(); t.bp_b = P(); t.ln_out_g = P(); t.ln_out_b = P();
    t.wz = P(); t.bz = P(); t.g_w = P(); t.g_b = P();
  };
  loadTM(tmo); loadTM(tmi);

  struct TA { const float *ln_g, *ln_b, *wb, *bb, *wq, *wk, *wv, *wo, *wg, *bg; } tas, tae;
  auto loadTA = [&](TA& t) {
    t.ln_g = P(); t.ln_b = P(); t.wb = P(); t.bb = P();
    t.wq = P(); t.wk = P(); t.wv = P(); t.wo = P(); t.wg = P(); t.bg = P();
  };
  loadTA(tas); loadTA(tae);

  const float* tp_ln_g = P(); const float* tp_ln_b = P();
  const float* tp_w1 = P(); const float* tp_b1 = P();
  const float* tp_w2 = P(); const float* tp_b2 = P();

  // ---- workspace arena -----------------------------------------------------
  char* wsb = (char*)d_ws;
  size_t off = 0;
  auto alloc = [&](size_t bytes) -> char* {
    char* r = wsb + off;
    off += (bytes + 255) & ~(size_t)255;
    return r;
  };
  auto castw = [&](const float* w, int n) -> const bf16* {
    bf16* d = (bf16*)alloc((size_t)n * 2);
    k_cast<<<(n + 255) / 256, 256, 0, stream>>>(w, d, n);
    return d;
  };

  // weights -> bf16
  const bf16* Wp2s = castw(p2s_w, 12 * 128);
  const bf16* Wq = castw(a_wq, 384 * 384);
  const bf16* Wk = castw(a_wk, 384 * 384);
  const bf16* Wv = castw(a_wv, 384 * 384);
  const bf16* Wo = castw(a_wo, 384 * 384);
  const bf16* Wg = castw(a_wg, 384 * 384);
  const bf16* Wt1 = castw(ts_w1, 1536 * 384);
  const bf16* Wt2 = castw(ts_w2, 384 * 1536);
  const bf16* Wsp = castw(sp_wp, 128 * 384);
  const bf16* Wspo = castw(sp_wo, 128 * 128);
  const bf16 *WtmA[2][6];
  const TM* tms[2] = { &tmo, &tmi };
  for (int t = 0; t < 2; ++t) {
    WtmA[t][0] = castw(tms[t]->ag_w, 128 * 128);
    WtmA[t][1] = castw(tms[t]->ap_w, 128 * 128);
    WtmA[t][2] = castw(tms[t]->bg_w, 128 * 128);
    WtmA[t][3] = castw(tms[t]->bp_w, 128 * 128);
    WtmA[t][4] = castw(tms[t]->wz, 128 * 128);
    WtmA[t][5] = castw(tms[t]->g_w, 128 * 128);
  }
  const bf16 *WtaA[2][6];
  const TA* tass[2] = { &tas, &tae };
  for (int t = 0; t < 2; ++t) {
    WtaA[t][0] = castw(tass[t]->wb, 4 * 128);
    WtaA[t][1] = castw(tass[t]->wq, 128 * 128);
    WtaA[t][2] = castw(tass[t]->wk, 128 * 128);
    WtaA[t][3] = castw(tass[t]->wv, 128 * 128);
    WtaA[t][4] = castw(tass[t]->wo, 128 * 128);
    WtaA[t][5] = castw(tass[t]->wg, 128 * 128);
  }
  const bf16* Wtp1 = castw(tp_w1, 512 * 128);
  const bf16* Wtp2 = castw(tp_w2, 128 * 512);

  // big buffers
  bf16* ZLN = (bf16*)alloc((size_t)LL * 128 * 2);
  bf16* BQ  = (bf16*)alloc((size_t)LL * 128 * 2);
  bf16* BK  = (bf16*)alloc((size_t)LL * 128 * 2);
  bf16* BVT = (bf16*)alloc((size_t)LL * 128 * 2);
  float* F1 = (float*)alloc((size_t)LL * 128 * 4);  // F1,F2 contiguous (64MB)
  float* F2 = (float*)alloc((size_t)LL * 128 * 4);
  float* TB = (float*)alloc((size_t)LL * 4 * 4);
  float* SBIAS = (float*)alloc((size_t)LL * 12 * 4);
  float* pm = (float*)alloc((size_t)LL * 4);
  float* madd_tri = (float*)alloc((size_t)LL * 4);
  float* madd_seq = (float*)alloc(256 * 4);
  bf16* SLN = (bf16*)alloc(256 * 384 * 2);
  bf16* SQ  = (bf16*)alloc(256 * 384 * 2);
  bf16* SK  = (bf16*)alloc(256 * 384 * 2);
  bf16* SVT = (bf16*)alloc(12 * 32 * 256 * 2);
  bf16* SY  = (bf16*)alloc(256 * 384 * 2);
  bf16* SH  = (bf16*)alloc(256 * 1536 * 2);
  float* SVF = (float*)alloc(256 * 384 * 4);
  float* SGATE = (float*)alloc(256 * 384 * 4);
  float* SS = (float*)alloc(256 * 128 * 4);

  float* seq  = (float*)d_out;         // [256,384]
  float* pair = seq + 256 * 384;       // [LL,128]
  hipMemcpyAsync(seq, in_seq, (size_t)256 * 384 * 4, hipMemcpyDeviceToDevice, stream);
  hipMemcpyAsync(pair, in_pair, (size_t)LL * 128 * 4, hipMemcpyDeviceToDevice, stream);

  const float kScale = 0.17677669529663688f;  // 1/sqrt(32)

  auto gemm = [&](const bf16* A, const bf16* W, int M, int N, int K,
                  const float* bias, const float* gate, const float* rowscale,
                  const float* res, float* outf, bf16* outb, int flags) {
    dim3 g((N + 63) / 64, (M + 63) / 64);
    if ((N & 63) == 0)
      k_gemm_nt_full<<<g, 128, 0, stream>>>(A, W, M, N, K, bias, gate, rowscale,
                                            res, outf, outb, flags);
    else
      k_gemm_nt<<<g, 128, 0, stream>>>(A, W, M, N, K, bias, gate, rowscale, res,
                                       outf, outb, flags);
  };
  auto lnorm = [&](const float* x, const float* g, const float* b, bf16* out,
                   int M, int D, int tr) {
    k_layernorm<<<M, 128, 0, stream>>>(x, g, b, out, D, tr);
  };

  // ---- mask prep
  k_prep_mask<<<256, 256, 0, stream>>>(in_mask, madd_seq, pm, madd_tri);

  // ---- pair->seq bias (from input pair, before pair updates)
  lnorm(pair, p2s_ln_g, p2s_ln_b, ZLN, LL, 128, 0);
  gemm(ZLN, Wp2s, LL, 12, 128, nullptr, nullptr, nullptr, nullptr, SBIAS, nullptr,
       F_OUTF);

  // ---- sequence self-attention with pair bias
  lnorm(seq, ln_seq_g, ln_seq_b, SLN, 256, 384, 0);
  gemm(SLN, Wq, 256, 384, 384, nullptr, nullptr, nullptr, nullptr, nullptr, SQ, F_OUTB);
  gemm(SLN, Wk, 256, 384, 384, nullptr, nullptr, nullptr, nullptr, nullptr, SK, F_OUTB);
  gemm(SLN, Wv, 256, 384, 384, nullptr, nullptr, nullptr, nullptr, SVF, nullptr, F_OUTF);
  gemm(SLN, Wg, 256, 384, 384, a_bg, nullptr, nullptr, nullptr, SGATE, nullptr,
       F_BIAS | F_SIG | F_OUTF);
  k_pack_vt<<<(12 * 32 * 256) / 256, 256, 0, stream>>>(SVF, SVT, 12, 384);
  k_attn<<<dim3(8, 12), 64, 0, stream>>>(SQ, SK, SVT, SBIAS, 12, SGATE, madd_seq, 0,
                                         SY, 12, 384, kScale);
  gemm(SY, Wo, 256, 384, 384, nullptr, nullptr, nullptr, seq, seq, nullptr,
       F_RES | F_OUTF);

  // ---- sequence transition
  lnorm(seq, ts_ln_g, ts_ln_b, SLN, 256, 384, 0);
  gemm(SLN, Wt1, 256, 1536, 384, ts_b1, nullptr, nullptr, nullptr, nullptr, SH,
       F_BIAS | F_RELU | F_OUTB);
  gemm(SH, Wt2, 256, 384, 1536, ts_b2, nullptr, nullptr, seq, seq, nullptr,
       F_BIAS | F_RES | F_OUTF);

  // ---- seq -> pair outer product
  lnorm(seq, sp_ln_g, sp_ln_b, SLN, 256, 384, 0);
  gemm(SLN, Wsp, 256, 128, 384, sp_bp, nullptr, nullptr, nullptr, SS, nullptr,
       F_BIAS | F_OUTF);
  k_s2p_outer<<<(LL * 128) / 256, 256, 0, stream>>>(SS, BQ);
  gemm(BQ, Wspo, LL, 128, 128, sp_bo, nullptr, nullptr, pair, pair, nullptr,
       F_BIAS | F_RES | F_OUTF);

  // ---- triangle multiplications (outgoing, incoming)
  for (int t = 0; t < 2; ++t) {
    const TM& tm = *tms[t];
    int incoming = t;  // 0 = outgoing, 1 = incoming
    lnorm(pair, tm.ln_in_g, tm.ln_in_b, ZLN, LL, 128, 0);
    gemm(ZLN, WtmA[t][0], LL, 128, 128, tm.ag_b, nullptr, nullptr, nullptr, F1,
         nullptr, F_BIAS | F_SIG | F_OUTF);
    gemm(ZLN, WtmA[t][1], LL, 128, 128, tm.ap_b, F1, pm, nullptr, F2, nullptr,
         F_BIAS | F_GATE | F_ROWSCALE | F_OUTF);
    k_pack_dlk<<<32768, 256, 0, stream>>>(F2, BQ, incoming);
    gemm(ZLN, WtmA[t][2], LL, 128, 128, tm.bg_b, nullptr, nullptr, nullptr, F1,
         nullptr, F_BIAS | F_SIG | F_OUTF);
    gemm(ZLN, WtmA[t][3], LL, 128, 128, tm.bp_b, F1, pm, nullptr, F2, nullptr,
         F_BIAS | F_GATE | F_ROWSCALE | F_OUTF);
    k_pack_dlk<<<32768, 256, 0, stream>>>(F2, BK, incoming);
    gemm(ZLN, WtmA[t][5], LL, 128, 128, tm.g_b, nullptr, nullptr, nullptr, F1,
         nullptr, F_BIAS | F_SIG | F_OUTF);
    k_einsum_nt<<<dim3(4, 4, 128), 128, 0, stream>>>(BQ, BK, F2);
    lnorm(F2, tm.ln_out_g, tm.ln_out_b, BVT, LL, 128, 0);
    gemm(BVT, WtmA[t][4], LL, 128, 128, tm.bz, F1, nullptr, pair, pair, nullptr,
         F_BIAS | F_GATE | F_RES | F_OUTF);
  }

  // ---- triangle attentions (starting, ending)
  for (int t = 0; t < 2; ++t) {
    const TA& ta = *tass[t];
    int ending = t;  // 0 = starting, 1 = ending (operate in transposed space)
    lnorm(pair, ta.ln_g, ta.ln_b, ZLN, LL, 128, ending);
    gemm(ZLN, WtaA[t][1], LL, 128, 128, nullptr, nullptr, nullptr, nullptr, nullptr,
         BQ, F_OUTB);
    gemm(ZLN, WtaA[t][2], LL, 128, 128, nullptr, nullptr, nullptr, nullptr, nullptr,
         BK, F_OUTB);
    gemm(ZLN, WtaA[t][3], LL, 128, 128, nullptr, nullptr, nullptr, nullptr, F2,
         nullptr, F_OUTF);
    k_pack_vt<<<32768, 256, 0, stream>>>(F2, BVT, 4, 128);
    gemm(ZLN, WtaA[t][5], LL, 128, 128, ta.bg, nullptr, nullptr, nullptr, F1,
         nullptr, F_BIAS | F_SIG | F_OUTF);
    gemm(ZLN, WtaA[t][0], LL, 4, 128, ta.bb, nullptr, nullptr, nullptr, TB, nullptr,
         F_BIAS | F_OUTF);
    // reuse ZLN as gated attention output (z no longer needed)
    k_attn<<<dim3(8, 1024), 64, 0, stream>>>(BQ, BK, BVT, TB, 4, F1, madd_tri, 256,
                                             ZLN, 4, 128, kScale);
    if (!ending) {
      gemm(ZLN, WtaA[t][4], LL, 128, 128, nullptr, nullptr, nullptr, pair, pair,
           nullptr, F_RES | F_OUTF);
    } else {
      gemm(ZLN, WtaA[t][4], LL, 128, 128, nullptr, nullptr, nullptr, nullptr, F2,
           nullptr, F_OUTF);
      k_add_transposed<<<32768, 256, 0, stream>>>(pair, F2);
    }
  }

  // ---- pair transition (hidden 64MB bf16 reuses contiguous F1+F2)
  lnorm(pair, tp_ln_g, tp_ln_b, ZLN, LL, 128, 0);
  bf16* H = (bf16*)F1;
  gemm(ZLN, Wtp1, LL, 512, 128, tp_b1, nullptr, nullptr, nullptr, nullptr, H,
       F_BIAS | F_RELU | F_OUTB);
  gemm(H, Wtp2, LL, 128, 512, tp_b2, nullptr, nullptr, pair, pair, nullptr,
       F_BIAS | F_RES | F_OUTF);
}
